// ODERN_Encoder_79104707657780
// MI455X (gfx1250) — matrix operand order repaired and checked by reading, not compiled, not run on hardware
//
#include <hip/hip_runtime.h>
#include <hip/hip_bf16.h>
#include <math.h>

typedef _Float16 half_t;
typedef __attribute__((ext_vector_type(16))) _Float16 v16h;
typedef __attribute__((ext_vector_type(8)))  _Float16 v8h;
typedef __attribute__((ext_vector_type(8)))  float    v8f;

#define HID 16
#define WID 64
#define LSEQ 256
#define BATCH 512
#define MAX_STEPS 12

// Tsit5 tableau. Row s = coefficients building stage-input s (s=0..6); row 6 = B (y_new).
static constexpr float TAB[7][6] = {
  {0.f,0.f,0.f,0.f,0.f,0.f},
  {0.161f,0.f,0.f,0.f,0.f,0.f},
  {-0.008480655492356989f, 0.335480655492357f, 0.f,0.f,0.f,0.f},
  {2.8971530571054935f, -6.359448489975075f, 4.3622954328695815f, 0.f,0.f,0.f},
  {5.325864828439257f, -11.748883564062828f, 7.4955393428898365f, -0.09249506636175525f, 0.f,0.f},
  {5.86145544294642f, -12.92096931784711f, 8.159367898576159f, -0.071584973281401f, -0.028269050394068383f, 0.f},
  {0.09646076681806523f, 0.01f, 0.4798896504144996f, 1.379008574103742f, -3.290069515436081f, 2.324710524099774f},
};
static constexpr float ETAB[7] = {
  -0.00178001105222577714f, -0.0008164344596567469f, 0.007880878010261995f,
  -0.1447110071732629f, 0.5823571654525552f, -0.45808210592918697f, 0.015151515151515152f,
};

#define LOG2E 1.4426950408889634f

__device__ __forceinline__ float rcp_fast(float x)  { return __builtin_amdgcn_rcpf(x); }
__device__ __forceinline__ float exp2_fast(float x) { return __builtin_amdgcn_exp2f(x); }
__device__ __forceinline__ float log2_fast(float x) { return __builtin_amdgcn_logf(x); }
__device__ __forceinline__ float sqrt_fast(float x) { return __builtin_amdgcn_sqrtf(x); }

// Branchless tanh: tanh(x) = sign(x) * (1-e)/(1+e), e = exp(-2|x|).  e in (0,1] -> no overflow.
__device__ __forceinline__ float tanh_fast(float x) {
  float e = exp2_fast(-2.f * LOG2E * fabsf(x));
  float r = (1.f - e) * rcp_fast(1.f + e);
  return copysignf(r, x);
}
// Branchless sigmoid: large -x -> exp2 -> +inf -> rcp -> 0.
__device__ __forceinline__ float sigmoid_fast(float x) {
  return rcp_fast(1.f + exp2_fast(-LOG2E * x));
}

__device__ __forceinline__ v8f wmma_f16(v16h a, v16h b, v8f c) {
  // v_wmma_f32_16x16x32_f16 : D = A(16x32 f16) * B(32x16 f16) + C(16x16 f32)
  return __builtin_amdgcn_wmma_f32_16x16x32_f16(false, a, false, b, (short)0, c, false, false);
}

// A operand from row-major f16 staging [16][rowK] (rowK multiple of 32, zero-padded).
// ISA 16-bit A layout: lane (m = lane&15, grp = lane>>4); element e: K = (e/8)*16 + grp*8 + (e&7) (+ kc*32).
__device__ __forceinline__ v16h load_a(const half_t* st, int rowK, int kc, int m, int grp) {
  const half_t* base = st + m * rowK + kc * 32 + grp * 8;
  v8h lo = *(const v8h*)(base);
  v8h hi = *(const v8h*)(base + 16);
  return __builtin_shufflevector(lo, hi, 0,1,2,3,4,5,6,7,8,9,10,11,12,13,14,15);
}

// B operand for y @ W^T from row-major f16 W[col][k] (rowlen multiple of 32, zero-padded).
// ISA 16-bit B layout: lane (n = lane&15, grp = lane>>4); element e: K = (e/8)*16 + grp*8 + (e&7) (+ kc*32).
__device__ __forceinline__ v16h load_b(const half_t* w, int rowlen, int colIdx, int kc, int grp) {
  const half_t* base = w + colIdx * rowlen + kc * 32 + grp * 8;
  v8h lo = *(const v8h*)(base);
  v8h hi = *(const v8h*)(base + 16);
  return __builtin_shufflevector(lo, hi, 0,1,2,3,4,5,6,7,8,9,10,11,12,13,14,15);
}

// One MLP evaluation: z[16x16 D-layout, 8 f32/lane] -> kout (16x16 D-layout).
__device__ __forceinline__ void mlp_eval(const float* z, float* kout,
    half_t* stZ, half_t* stA, half_t* stB,
    const half_t* sW0, const half_t* sW1, const half_t* sW2, const half_t* sW3,
    const float* bias0, const float* bias1, const float* bias2, float bias3,
    int col, int grp)
{
  // z (D layout: row = grp*8+v, col = col) -> zero-padded staging [16][32] for A operand
  #pragma unroll
  for (int v = 0; v < 8; ++v)
    stZ[(grp * 8 + v) * 32 + col] = (half_t)z[v];
  __syncthreads();
  v16h aZ = load_a(stZ, 32, 0, col, grp);
  __syncthreads();

  // layer0: [16x16] @ [16x64], K padded 16->32 with zeros (in sW0 and stZ)
  #pragma unroll
  for (int n = 0; n < 4; ++n) {
    v8f acc = {};
    acc = wmma_f16(aZ, load_b(sW0, 32, n * 16 + col, 0, grp), acc);
    #pragma unroll
    for (int v = 0; v < 8; ++v)
      stB[(grp * 8 + v) * WID + n * 16 + col] = (half_t)tanh_fast(acc[v] + bias0[n]);
  }
  __syncthreads();

  // layer1: [16x64] @ [64x64]
  v16h a1k0 = load_a(stB, WID, 0, col, grp);
  v16h a1k1 = load_a(stB, WID, 1, col, grp);
  __syncthreads();
  #pragma unroll
  for (int n = 0; n < 4; ++n) {
    v8f acc = {};
    acc = wmma_f16(a1k0, load_b(sW1, WID, n * 16 + col, 0, grp), acc);
    acc = wmma_f16(a1k1, load_b(sW1, WID, n * 16 + col, 1, grp), acc);
    #pragma unroll
    for (int v = 0; v < 8; ++v)
      stA[(grp * 8 + v) * WID + n * 16 + col] = (half_t)tanh_fast(acc[v] + bias1[n]);
  }
  __syncthreads();

  // layer2: [16x64] @ [64x64]
  v16h a2k0 = load_a(stA, WID, 0, col, grp);
  v16h a2k1 = load_a(stA, WID, 1, col, grp);
  __syncthreads();
  #pragma unroll
  for (int n = 0; n < 4; ++n) {
    v8f acc = {};
    acc = wmma_f16(a2k0, load_b(sW2, WID, n * 16 + col, 0, grp), acc);
    acc = wmma_f16(a2k1, load_b(sW2, WID, n * 16 + col, 1, grp), acc);
    #pragma unroll
    for (int v = 0; v < 8; ++v)
      stB[(grp * 8 + v) * WID + n * 16 + col] = (half_t)tanh_fast(acc[v] + bias2[n]);
  }
  __syncthreads();

  // layer3: [16x64] @ [64x16] (no tanh)
  v16h a3k0 = load_a(stB, WID, 0, col, grp);
  v16h a3k1 = load_a(stB, WID, 1, col, grp);
  v8f acc = {};
  acc = wmma_f16(a3k0, load_b(sW3, WID, col, 0, grp), acc);
  acc = wmma_f16(a3k1, load_b(sW3, WID, col, 1, grp), acc);
  #pragma unroll
  for (int v = 0; v < 8; ++v) kout[v] = acc[v] + bias3;
  __syncthreads();
}

extern "C" __global__ __launch_bounds__(32)
void odernn_encoder_kernel(const float* __restrict__ x_seq,
    const float* __restrict__ W0, const float* __restrict__ b0,
    const float* __restrict__ W1, const float* __restrict__ b1,
    const float* __restrict__ W2, const float* __restrict__ b2,
    const float* __restrict__ W3, const float* __restrict__ b3,
    const float* __restrict__ Wih, const float* __restrict__ Whh,
    const float* __restrict__ bih, const float* __restrict__ bnv,
    const float* __restrict__ Wp, const float* __restrict__ bp,
    float* __restrict__ out)
{
  __shared__ __attribute__((aligned(16))) half_t sW0[WID * 32];        // [64][32] K-padded
  __shared__ __attribute__((aligned(16))) half_t sW1[WID * WID];       // [64][64]
  __shared__ __attribute__((aligned(16))) half_t sW2[WID * WID];       // [64][64]
  __shared__ __attribute__((aligned(16))) half_t sW3[HID * WID];       // [16][64]
  __shared__ __attribute__((aligned(16))) half_t sWhh[3 * HID * 32];   // [48][32] K-padded
  __shared__ __attribute__((aligned(16))) half_t stZ[16 * 32];         // [16][32] K-padded A staging
  __shared__ __attribute__((aligned(16))) half_t stA[16 * WID];        // staging
  __shared__ __attribute__((aligned(16))) half_t stB[16 * WID];        // staging

  const int lane = (int)(threadIdx.x & 31u);
  const int col  = lane & 15;   // N for B/C/D tiles, M for A tiles
  const int grp  = lane >> 4;
  const int br   = (int)blockIdx.x * 16;

  // Stage weights to LDS as f16; zero-pad K where real K=16.
  for (int i = lane; i < WID * 32; i += 32) {
    int o = i >> 5, k = i & 31;
    sW0[i] = (k < HID) ? (half_t)W0[o * HID + k] : (half_t)0.f;
  }
  for (int i = lane; i < WID * WID; i += 32) { sW1[i] = (half_t)W1[i]; sW2[i] = (half_t)W2[i]; }
  for (int i = lane; i < HID * WID; i += 32) sW3[i] = (half_t)W3[i];
  for (int i = lane; i < 3 * HID * 32; i += 32) {
    int o = i >> 5, k = i & 31;
    sWhh[i] = (k < HID) ? (half_t)Whh[o * HID + k] : (half_t)0.f;
  }
  for (int i = lane; i < 16 * 32; i += 32) stZ[i] = (half_t)0.f;  // pad region stays 0 forever
  __syncthreads();

  // Per-lane bias / small-weight copies (indexed by output column = col).
  float bias0[4], bias1[4], bias2[4];
  #pragma unroll
  for (int n = 0; n < 4; ++n) {
    bias0[n] = b0[n * 16 + col];
    bias1[n] = b1[n * 16 + col];
    bias2[n] = b2[n * 16 + col];
  }
  const float bias3 = b3[col];
  float wihx[3], wihy[3], bih_l[3];
  #pragma unroll
  for (int n = 0; n < 3; ++n) {
    wihx[n]  = Wih[(n * 16 + col) * 2 + 0];
    wihy[n]  = Wih[(n * 16 + col) * 2 + 1];
    bih_l[n] = bih[n * 16 + col];
  }
  const float bn_l = bnv[col];
  const float wp_l = Wp[col];
  const float bp_l = bp[0];

  // Hidden state in D layout: y[v] = h[row = grp*8+v][col]
  float y[8];
  #pragma unroll
  for (int v = 0; v < 8; ++v) y[v] = 0.f;

  #pragma unroll 1
  for (int step = 0; step < LSEQ; ++step) {
    // ---------------- adaptive Tsit5 ODE solve over t in [0,1] ----------------
    float t[8], dtv[8];
    #pragma unroll
    for (int v = 0; v < 8; ++v) { t[v] = 0.f; dtv[v] = 1.f; }

    #pragma unroll 1
    for (int it = 0; it < MAX_STEPS; ++it) {
      float dtc[8]; bool done[8];
      #pragma unroll
      for (int v = 0; v < 8; ++v) {
        done[v] = t[v] >= (1.0f - 1e-7f);
        dtc[v] = done[v] ? 1.0f : fminf(dtv[v], 1.0f - t[v]);
      }

      float k[7][8];
      float ynew[8];
      #pragma unroll
      for (int s = 0; s < 7; ++s) {
        float z[8];
        #pragma unroll
        for (int v = 0; v < 8; ++v) {
          float acc = 0.f;
          #pragma unroll
          for (int j = 0; j < 6; ++j)
            if (j < s) acc += TAB[s][j] * k[j][v];
          z[v] = y[v] + dtc[v] * acc;
        }
        if (s == 6) {
          #pragma unroll
          for (int v = 0; v < 8; ++v) ynew[v] = z[v];
        }
        mlp_eval(z, k[s], stZ, stA, stB, sW0, sW1, sW2, sW3,
                 bias0, bias1, bias2, bias3, col, grp);
      }

      // error norm: mean over 16 hidden columns (16-lane shuffle reduction)
      float se[8];
      #pragma unroll
      for (int v = 0; v < 8; ++v) {
        float ye = 0.f;
        #pragma unroll
        for (int j = 0; j < 7; ++j) ye += ETAB[j] * k[j][v];
        ye *= dtc[v];
        float sc = 1e-4f + 1e-2f * fmaxf(fabsf(y[v]), fabsf(ynew[v]));
        float r = ye * rcp_fast(sc);
        se[v] = r * r;
      }
      #pragma unroll
      for (int m = 1; m < 16; m <<= 1) {
        #pragma unroll
        for (int v = 0; v < 8; ++v) se[v] += __shfl_xor(se[v], m, 32);
      }
      #pragma unroll
      for (int v = 0; v < 8; ++v) {
        float err = sqrt_fast(se[v] * (1.0f / 16.0f));
        bool upd = (!done[v]) && (err <= 1.0f);
        y[v] = upd ? ynew[v] : y[v];
        t[v] = upd ? (t[v] + dtc[v]) : t[v];
        // err^-0.2 via exp2/log2 (err >= 1e-10 > 0)
        float f = 0.9f * exp2_fast(-0.2f * log2_fast(fmaxf(err, 1e-10f)));
        f = fminf(fmaxf(f, 0.2f), 10.0f);
        dtv[v] = done[v] ? dtv[v] : (dtc[v] * f);
      }
    }

    // ---------------- GRU observation update ----------------
    // hg = h_ev @ Whh^T via 3 WMMA tiles (K=16 zero-padded to 32)
    #pragma unroll
    for (int v = 0; v < 8; ++v)
      stZ[(grp * 8 + v) * 32 + col] = (half_t)y[v];
    __syncthreads();
    v16h ah = load_a(stZ, 32, 0, col, grp);
    __syncthreads();
    v8f hr = {}, hz = {}, hn = {};
    hr = wmma_f16(ah, load_b(sWhh, 32, 0 * 16 + col, 0, grp), hr);
    hz = wmma_f16(ah, load_b(sWhh, 32, 1 * 16 + col, 0, grp), hz);
    hn = wmma_f16(ah, load_b(sWhh, 32, 2 * 16 + col, 0, grp), hn);

    #pragma unroll
    for (int v = 0; v < 8; ++v) {
      const float* xp = x_seq + ((size_t)(br + grp * 8 + v) * LSEQ + (size_t)step) * 2;
      const float x0 = xp[0], x1 = xp[1];
      const float ir  = x0 * wihx[0] + x1 * wihy[0] + bih_l[0];
      const float iz  = x0 * wihx[1] + x1 * wihy[1] + bih_l[1];
      const float in_ = x0 * wihx[2] + x1 * wihy[2] + bih_l[2];
      const float r  = sigmoid_fast(ir + hr[v]);
      const float zg = sigmoid_fast(iz + hz[v]);
      const float nn = tanh_fast(in_ + r * (hn[v] + bn_l));
      y[v] = nn + zg * (y[v] - nn);
    }
    __syncthreads();
  }

  // ---------------- final projection out = hF @ Wp^T + bp ----------------
  float part[8];
  #pragma unroll
  for (int v = 0; v < 8; ++v) part[v] = y[v] * wp_l;
  #pragma unroll
  for (int m = 1; m < 16; m <<= 1) {
    #pragma unroll
    for (int v = 0; v < 8; ++v) part[v] += __shfl_xor(part[v], m, 32);
  }
  if (col == 0) {
    #pragma unroll
    for (int v = 0; v < 8; ++v)
      out[br + grp * 8 + v] = part[v] + bp_l;
  }
}

extern "C" void kernel_launch(void* const* d_in, const int* in_sizes, int n_in,
                              void* d_out, int out_size, void* d_ws, size_t ws_size,
                              hipStream_t stream) {
  (void)in_sizes; (void)n_in; (void)out_size; (void)d_ws; (void)ws_size;
  const float* x_seq = (const float*)d_in[0];
  const float* W0  = (const float*)d_in[1];
  const float* b0  = (const float*)d_in[2];
  const float* W1  = (const float*)d_in[3];
  const float* b1  = (const float*)d_in[4];
  const float* W2  = (const float*)d_in[5];
  const float* b2  = (const float*)d_in[6];
  const float* W3  = (const float*)d_in[7];
  const float* b3  = (const float*)d_in[8];
  const float* Wih = (const float*)d_in[9];
  const float* Whh = (const float*)d_in[10];
  const float* bih = (const float*)d_in[11];
  const float* bn  = (const float*)d_in[12];
  const float* Wp  = (const float*)d_in[13];
  const float* bp  = (const float*)d_in[14];
  float* out = (float*)d_out;

  hipLaunchKernelGGL(odernn_encoder_kernel, dim3(BATCH / 16), dim3(32), 0, stream,
                     x_seq, W0, b0, W1, b1, W2, b2, W3, b3,
                     Wih, Whh, bih, bn, Wp, bp, out);
}
